// COSAL_33981781246135
// MI455X (gfx1250) — matrix operand order repaired and checked by reading, not compiled, not run on hardware
//
#include <hip/hip_runtime.h>

// ---------------------------------------------------------------------------
// Capsule-routing GNN for MI455X (gfx1250, wave32).
//  * PCA GEMM (100000x512 @ 512x512) on v_wmma_f32_16x16x32_bf16.
//    A tile staged once per block into LDS via global_load_async_to_lds_b128
//    (ASYNCcnt), K loop fully unrolled (no accumulator phi copies, no
//    barriers in the loop), fused bias+ReLU+per-capsule L2 norm epilogue.
//  * Routing: CSR scatter (no atomics for u), atomic segment sums for the
//    softmaxes (max-subtraction dropped: all softmax inputs bounded <= 1).
//  * MLP + log_softmax on WMMA with N padded to 48, fully unrolled.
// ---------------------------------------------------------------------------

typedef __bf16 bf16;
typedef __attribute__((ext_vector_type(16))) __bf16 v16bf;
typedef __attribute__((ext_vector_type(8)))  __bf16 v8bf;
typedef __attribute__((ext_vector_type(2)))  __bf16 v2bf;
typedef __attribute__((ext_vector_type(8)))  float  v8f;

union V16 { v16bf v; v8bf h[2]; };

#define KCAP   8
#define DDIM   512
#define TNODE  16384
#define NBN    100000
#define EEDGE  131072
#define FBETA  0.5f

// ---------------- weight -> bf16 WMMA fragment layout ----------------------
// frag(kt,nt): 32 lanes x 16 bf16, element i of lane l:
//   K = kt*32 + (l<16?0:16) + i ,  N = nt*16 + (l&15)
__global__ void cosal_wfrag_pca(const float* __restrict__ W, bf16* __restrict__ wf) {
  int idx = blockIdx.x * 256 + threadIdx.x;          // 16*32*32 threads
  if (idx >= 16 * 32 * 32) return;
  int kt = idx >> 10, rem = idx & 1023;
  int nt = rem >> 5, lane = rem & 31;
  int l16 = lane & 15;
  int koff = (lane & 16) ? 16 : 0;
  bf16* dst = wf + ((size_t)(kt * 32 + nt) * 32 + lane) * 16;
  #pragma unroll
  for (int i = 0; i < 16; ++i)
    dst[i] = (bf16)W[(size_t)(kt * 32 + koff + i) * DDIM + nt * 16 + l16];
}

__global__ void cosal_wfrag_mlp(const float* __restrict__ W, bf16* __restrict__ wf) {
  int idx = blockIdx.x * 256 + threadIdx.x;          // 16*3*32 threads
  if (idx >= 16 * 3 * 32) return;
  int kt = idx / 96, rem = idx % 96;
  int nt = rem / 32, lane = rem % 32;
  int l16 = lane & 15;
  int koff = (lane & 16) ? 16 : 0;
  int N = nt * 16 + l16;
  bf16* dst = wf + ((size_t)(kt * 3 + nt) * 32 + lane) * 16;
  #pragma unroll
  for (int i = 0; i < 16; ++i) {
    int K = kt * 32 + koff + i;
    dst[i] = (N < 40) ? (bf16)W[(size_t)K * 40 + N] : (bf16)0.0f;
  }
}

// ---------------- PCA GEMM + bias + ReLU + per-capsule L2 norm -------------
// block = 256 (8 waves); block owns 16 rows x 512 cols; wave w = capsule w
// (16 rows x 64 cols = 4 WMMA accumulators). A tile (16x512 f32, 32KB) staged
// once via async-to-LDS DMA; K loop fully unrolled, barrier-free.
__global__ void cosal_gemm_pca(const float* __restrict__ xnb,
                               const bf16* __restrict__ wfA,
                               const float* __restrict__ bias,
                               bf16* __restrict__ xn) {
  const int tid  = threadIdx.x;
  const int lane = tid & 31;
  const int wid  = tid >> 5;            // capsule
  const int l16  = lane & 15;
  const int half = lane >> 4;
  const int koff = half ? 16 : 0;
  const int m0   = blockIdx.x * 16;

  __shared__ float ldsAf[16 * 512];     // 32 KB A tile (f32)

  // ---- async DMA stage: 8 x b128 per thread, tracked by ASYNCcnt ----------
  {
    typedef __attribute__((address_space(3))) float ldsf_t;
    unsigned ldsbase = (unsigned)(unsigned long long)(ldsf_t*)ldsAf;
    #pragma unroll
    for (int q = 0; q < 8; ++q) {
      int idx = (q * 256 + tid) * 4;    // f32 element index, 16B-aligned
      int r = idx >> 9, kk = idx & 511;
      const float* gp = xnb + (size_t)(m0 + r) * DDIM + kk;
      unsigned la = ldsbase + (unsigned)idx * 4u;
      asm volatile("global_load_async_to_lds_b128 %0, %1, off"
                   :: "v"(la), "v"((unsigned long long)gp) : "memory");
    }
    asm volatile("s_wait_asynccnt 0x0" ::: "memory");
  }
  __syncthreads();

  v8f acc[4] = {};
  #pragma unroll
  for (int kt = 0; kt < 16; ++kt) {
    // A frag from LDS (layout: lanes<16 K{0..15}, lanes>=16
    // K{16..31}); cvt f32->bf16 in registers.
    const float* ap = ldsAf + l16 * DDIM + kt * 32 + koff;
    float4 f0 = *(const float4*)(ap + 0),  f1 = *(const float4*)(ap + 4);
    float4 f2 = *(const float4*)(ap + 8), f3 = *(const float4*)(ap + 12);
    float tv[16] = {f0.x,f0.y,f0.z,f0.w, f1.x,f1.y,f1.z,f1.w,
                    f2.x,f2.y,f2.z,f2.w, f3.x,f3.y,f3.z,f3.w};
    V16 A;
    #pragma unroll
    for (int i = 0; i < 16; ++i) A.v[i] = (bf16)tv[i];

    #pragma unroll
    for (int j = 0; j < 4; ++j) {
      v16bf b = ((const v16bf*)wfA)[(size_t)(kt * 32 + wid * 4 + j) * 32 + lane];
      acc[j] = __builtin_amdgcn_wmma_f32_16x16x32_bf16(
          false, A.v, false, b, (short)0, acc[j], false, false);
    }
  }

  // epilogue: bias + relu + per-capsule l2 norm (rows m0+r / m0+r+8)
  float sq[8];
  #pragma unroll
  for (int r = 0; r < 8; ++r) sq[r] = 0.f;
  #pragma unroll
  for (int j = 0; j < 4; ++j) {
    float bv = bias[wid * 64 + j * 16 + l16];
    #pragma unroll
    for (int r = 0; r < 8; ++r) {
      float v = fmaxf(acc[j][r] + bv, 0.f);
      acc[j][r] = v;
      sq[r] += v * v;
    }
  }
  #pragma unroll
  for (int r = 0; r < 8; ++r) {
    float s = sq[r];
    s += __shfl_xor(s, 1); s += __shfl_xor(s, 2);
    s += __shfl_xor(s, 4); s += __shfl_xor(s, 8);
    sq[r] = 1.0f / fmaxf(sqrtf(s), 1e-12f);
  }
  int rowbase = m0 + (half ? 8 : 0);
  #pragma unroll
  for (int j = 0; j < 4; ++j) {
    int colc = wid * 64 + j * 16 + l16;
    #pragma unroll
    for (int r = 0; r < 8; ++r)
      xn[(size_t)(rowbase + r) * DDIM + colc] = (bf16)(acc[j][r] * sq[r]);
  }
}

// ---------------- small helpers --------------------------------------------
__global__ void cosal_zero(float* __restrict__ p, int n) {
  int i = blockIdx.x * 256 + threadIdx.x;
  if (i < n) p[i] = 0.f;
}

__global__ void cosal_segptr(const int* __restrict__ row, int* __restrict__ ptr) {
  int t = blockIdx.x * 256 + threadIdx.x;
  if (t > TNODE) return;
  int lo = 0, hi = EEDGE;
  while (lo < hi) { int mid = (lo + hi) >> 1; if (row[mid] < t) lo = mid + 1; else hi = mid; }
  ptr[t] = lo;
}

// ppr segment softmax (bounded input -> no max pass)
__global__ void cosal_ppr_sum(const float* __restrict__ ppr, const int* __restrict__ row,
                              float* __restrict__ s0) {
  int e = blockIdx.x * 256 + threadIdx.x;
  if (e < EEDGE) atomicAdd(&s0[row[e]], expf(ppr[e]));
}
__global__ void cosal_ppr_norm(const float* __restrict__ ppr, const int* __restrict__ row,
                               const float* __restrict__ s0, float* __restrict__ pprn) {
  int e = blockIdx.x * 256 + threadIdx.x;
  if (e < EEDGE) pprn[e] = expf(ppr[e]) / s0[row[e]];
}

// ---------------- init u: CSR scatter of ppr-weighted z --------------------
__global__ void cosal_init_u(const bf16* __restrict__ xn, const int* __restrict__ col,
                             const int* __restrict__ ptr, const float* __restrict__ pprn,
                             float* __restrict__ u) {
  int t = blockIdx.x;
  int tid = threadIdx.x;
  int c = tid * 2;
  int beg = ptr[t], end = ptr[t + 1];
  float a0 = 0.f, a1 = 0.f;
  for (int e = beg; e < end; ++e) {
    float w = pprn[e];
    v2bf z = *(const v2bf*)(xn + (size_t)col[e] * DDIM + c);
    a0 += w * (float)z[0]; a1 += w * (float)z[1];
  }
  u[(size_t)t * DDIM + c] = a0;
  u[(size_t)t * DDIM + c + 1] = a1;
}

// ---------------- routing: edge capsule logits + first softmax numer -------
// one wave per edge (8 edges / 256-thread block); lane owns 16 dims,
// capsule k = lane/4, reduce over 4 lanes.
__global__ void cosal_logits(const float* __restrict__ u, const bf16* __restrict__ xn,
                             const int* __restrict__ row, const int* __restrict__ col,
                             float* __restrict__ e1, float* __restrict__ s1) {
  int wid = threadIdx.x >> 5, lane = threadIdx.x & 31;
  int e = blockIdx.x * 8 + wid;
  int t = row[e], cc = col[e];
  v16bf z = *(const v16bf*)(xn + (size_t)cc * DDIM + lane * 16);
  const float* up = u + (size_t)t * DDIM + lane * 16;
  float4 u0 = *(const float4*)(up + 0),  u1 = *(const float4*)(up + 4);
  float4 u2 = *(const float4*)(up + 8),  u3 = *(const float4*)(up + 12);
  float uu[16] = {u0.x,u0.y,u0.z,u0.w, u1.x,u1.y,u1.z,u1.w,
                  u2.x,u2.y,u2.z,u2.w, u3.x,u3.y,u3.z,u3.w};
  float s = 0.f;
  #pragma unroll
  for (int i = 0; i < 16; ++i) s += (float)z[i] * uu[i];
  s += __shfl_xor(s, 1); s += __shfl_xor(s, 2);
  if ((lane & 3) == 0) {
    int k = lane >> 2;
    float ev = expf(s);                    // |s| <= 1 -> safe w/o max shift
    e1[(size_t)e * 8 + k] = ev;
    atomicAdd(&s1[t * 8 + k], ev);
  }
}

// blend with ppr, second softmax numerators
__global__ void cosal_blend(const int* __restrict__ row, const float* __restrict__ pprn,
                            const float* __restrict__ e1, const float* __restrict__ s1,
                            float* __restrict__ e2, float* __restrict__ s2) {
  int idx = blockIdx.x * 256 + threadIdx.x;     // E*8 threads
  int e = idx >> 3, k = idx & 7;
  int t = row[e];
  float p2 = FBETA * (e1[idx] / s1[t * 8 + k]) + (1.f - FBETA) * pprn[e];
  float ev = expf(p2);                           // p2 in (0,1]
  e2[idx] = ev;
  atomicAdd(&s2[t * 8 + k], ev);
}

// u update: CSR scatter of p-weighted z, optional per-capsule l2 norm.
// 256 threads = 8 waves; wave k == capsule k (cols 64k..64k+63, 2/lane).
__global__ void cosal_update_u(const bf16* __restrict__ xn, const int* __restrict__ col,
                               const int* __restrict__ ptr, const float* __restrict__ e2,
                               const float* __restrict__ s2, float* __restrict__ u,
                               int do_norm) {
  int t = blockIdx.x;
  int tid = threadIdx.x;
  int k = tid >> 5;
  int c = tid * 2;
  int beg = ptr[t], end = ptr[t + 1];
  float invs = (end > beg) ? 1.0f / s2[t * 8 + k] : 0.f;
  float a0 = 0.f, a1 = 0.f;
  for (int e = beg; e < end; ++e) {
    float w = e2[(size_t)e * 8 + k] * invs;
    v2bf z = *(const v2bf*)(xn + (size_t)col[e] * DDIM + c);
    a0 += w * (float)z[0]; a1 += w * (float)z[1];
  }
  if (do_norm) {
    float sq = a0 * a0 + a1 * a1;
    sq += __shfl_xor(sq, 1);  sq += __shfl_xor(sq, 2);  sq += __shfl_xor(sq, 4);
    sq += __shfl_xor(sq, 8);  sq += __shfl_xor(sq, 16);
    float sc = 1.0f / fmaxf(sqrtf(sq), 1e-12f);
    a0 *= sc; a1 *= sc;
  }
  u[(size_t)t * DDIM + c] = a0;
  u[(size_t)t * DDIM + c + 1] = a1;
}

// ---------------- MLP head on WMMA + row log_softmax -----------------------
// 8 waves/block, wave owns a 16-row tile x 48 cols (3 accumulators, N padded).
__global__ void cosal_mlp(const float* __restrict__ u, const bf16* __restrict__ wfM,
                          const float* __restrict__ bias, float* __restrict__ out) {
  const int lane = threadIdx.x & 31;
  const int wid  = threadIdx.x >> 5;
  const int l16  = lane & 15;
  const int half = lane >> 4;
  const int koff = half ? 16 : 0;
  const int m0   = (blockIdx.x * 8 + wid) * 16;

  v8f acc[3] = {};
  #pragma unroll
  for (int kt = 0; kt < 16; ++kt) {
    const float* src = u + (size_t)(m0 + l16) * DDIM + kt * 32 + koff;
    float4 f0 = *(const float4*)(src + 0),  f1 = *(const float4*)(src + 4);
    float4 f2 = *(const float4*)(src + 8), f3 = *(const float4*)(src + 12);
    float tv[16] = {f0.x,f0.y,f0.z,f0.w, f1.x,f1.y,f1.z,f1.w,
                    f2.x,f2.y,f2.z,f2.w, f3.x,f3.y,f3.z,f3.w};
    V16 A;
    #pragma unroll
    for (int i = 0; i < 16; ++i) A.v[i] = (bf16)fmaxf(tv[i], 0.f);   // relu(u)
    #pragma unroll
    for (int j = 0; j < 3; ++j) {
      v16bf b = ((const v16bf*)wfM)[(size_t)(kt * 3 + j) * 32 + lane];
      acc[j] = __builtin_amdgcn_wmma_f32_16x16x32_bf16(
          false, A.v, false, b, (short)0, acc[j], false, false);
    }
  }

  // bias (cols < 40) then row-wise log_softmax via half-wave reductions
  #pragma unroll
  for (int j = 0; j < 3; ++j) {
    int n = j * 16 + l16;
    float bv = (n < 40) ? bias[n] : 0.f;
    #pragma unroll
    for (int r = 0; r < 8; ++r) acc[j][r] += bv;
  }
  bool v2ok = (l16 < 8);
  #pragma unroll
  for (int r = 0; r < 8; ++r) {
    float m = fmaxf(acc[0][r], acc[1][r]);
    if (v2ok) m = fmaxf(m, acc[2][r]);
    m = fmaxf(m, __shfl_xor(m, 1)); m = fmaxf(m, __shfl_xor(m, 2));
    m = fmaxf(m, __shfl_xor(m, 4)); m = fmaxf(m, __shfl_xor(m, 8));
    float se = expf(acc[0][r] - m) + expf(acc[1][r] - m)
             + (v2ok ? expf(acc[2][r] - m) : 0.f);
    se += __shfl_xor(se, 1); se += __shfl_xor(se, 2);
    se += __shfl_xor(se, 4); se += __shfl_xor(se, 8);
    float L = m + logf(se);
    int rw = m0 + r + (half ? 8 : 0);
    out[(size_t)rw * 40 + l16]      = acc[0][r] - L;
    out[(size_t)rw * 40 + 16 + l16] = acc[1][r] - L;
    if (v2ok) out[(size_t)rw * 40 + 32 + l16] = acc[2][r] - L;
  }
}

// ---------------------------------------------------------------------------
extern "C" void kernel_launch(void* const* d_in, const int* in_sizes, int n_in,
                              void* d_out, int out_size, void* d_ws, size_t ws_size,
                              hipStream_t stream) {
  (void)in_sizes; (void)n_in; (void)out_size; (void)ws_size;
  const float* x_nb  = (const float*)d_in[0];
  const float* ppr   = (const float*)d_in[1];
  const float* pca_w = (const float*)d_in[2];
  const float* pca_b = (const float*)d_in[3];
  const float* mlp_w = (const float*)d_in[4];
  const float* mlp_b = (const float*)d_in[5];
  const int*   row   = (const int*)d_in[6];
  const int*   col   = (const int*)d_in[7];
  // d_in[8] (x_idx) unused by the reference computation.
  float* out = (float*)d_out;

  char* base = (char*)d_ws;
  size_t off = 0;
  auto take = [&](size_t bytes) -> char* {
    char* p = base + off;
    off = (off + bytes + 255) & ~(size_t)255;
    return p;
  };
  bf16*  wfA  = (bf16*) take((size_t)512 * 512 * 2);          // pca_w frags
  bf16*  wfM  = (bf16*) take((size_t)16 * 3 * 32 * 16 * 2);   // mlp_w frags (N pad 48)
  bf16*  xn   = (bf16*) take((size_t)NBN * DDIM * 2);         // normalized x (bf16)
  float* pprn = (float*)take((size_t)EEDGE * 4);
  float* s0   = (float*)take((size_t)TNODE * 4);
  int*   ptr  = (int*)  take((size_t)(TNODE + 1) * 4);
  float* u    = (float*)take((size_t)TNODE * DDIM * 4);
  float* e1   = (float*)take((size_t)EEDGE * 8 * 4);
  float* e2   = (float*)take((size_t)EEDGE * 8 * 4);
  float* s1   = (float*)take((size_t)TNODE * 8 * 4 * 2);      // s1 | s2 contiguous
  float* s2   = s1 + (size_t)TNODE * 8;

  // weight fragment conversion
  cosal_wfrag_pca<<<64, 256, 0, stream>>>(pca_w, wfA);
  cosal_wfrag_mlp<<<6, 256, 0, stream>>>(mlp_w, wfM);

  // ppr segment softmax
  cosal_zero<<<(TNODE + 255) / 256, 256, 0, stream>>>(s0, TNODE);
  cosal_ppr_sum<<<EEDGE / 256, 256, 0, stream>>>(ppr, row, s0);
  cosal_ppr_norm<<<EEDGE / 256, 256, 0, stream>>>(ppr, row, s0, pprn);

  // CSR pointers from sorted row_idx
  cosal_segptr<<<(TNODE + 1 + 255) / 256, 256, 0, stream>>>(row, ptr);

  // PCA GEMM + relu + capsule l2norm  (NB = 6250 * 16 exactly)
  cosal_gemm_pca<<<NBN / 16, 256, 0, stream>>>(x_nb, wfA, pca_b, xn);

  // init u = segment_sum(ppr * z)
  cosal_init_u<<<TNODE, 256, 0, stream>>>(xn, col, ptr, pprn, u);

  // 3 routing iterations
  for (int it = 0; it < 3; ++it) {
    cosal_zero<<<(TNODE * 16) / 256, 256, 0, stream>>>(s1, TNODE * 16);  // s1+s2
    cosal_logits<<<EEDGE / 8, 256, 0, stream>>>(u, xn, row, col, e1, s1);
    cosal_blend<<<(EEDGE * 8) / 256, 256, 0, stream>>>(row, pprn, e1, s1, e2, s2);
    cosal_update_u<<<TNODE, 256, 0, stream>>>(xn, col, ptr, e2, s2, u, it < 2 ? 1 : 0);
  }

  // MLP head + log_softmax (1024 row-tiles, 8 per block)
  cosal_mlp<<<TNODE / 16 / 8, 256, 0, stream>>>(u, wfM, mlp_b, out);
}
